// Encoder_31971736552215
// MI455X (gfx1250) — matrix operand order repaired and checked by reading, not compiled, not run on hardware
//
#include <hip/hip_runtime.h>
#include <hip/hip_bf16.h>

// ---------------------------------------------------------------------------
// Problem sizes (from reference)
// ---------------------------------------------------------------------------
#define N_NODES 60000
#define N_CASC  32
#define N_IN    64
#define N_HID   128
#define N_OUT   64

typedef __attribute__((ext_vector_type(16))) _Float16 v16h;
typedef __attribute__((ext_vector_type(8)))  float    v8f;
typedef __attribute__((ext_vector_type(4)))  int      v4i_;

union Frag {
    uint4 q[2];
    v16h  v;
};

// f32 -> f16 (RTE via hardware cvt), packed pair into one dword
__device__ __forceinline__ unsigned short f2h(float x) {
    union { _Float16 h; unsigned short s; } u;
    u.h = (_Float16)x;
    return u.s;
}
__device__ __forceinline__ unsigned pkh(float a, float b) {
    return (unsigned)f2h(a) | ((unsigned)f2h(b) << 16);
}

// ---------------------------------------------------------------------------
// CDNA5 async global->LDS staging (ASYNCcnt path), with sync fallback
// ---------------------------------------------------------------------------
#if __has_builtin(__builtin_amdgcn_global_load_async_to_lds_b128)
#define ASYNC_STAGE 1
#else
#define ASYNC_STAGE 0
#endif

#define GLOBAL_V4I(p) ((__attribute__((address_space(1))) v4i_*)(p))
#define LDS_V4I(p)    ((__attribute__((address_space(3))) v4i_*)(p))

// Stage one cascade's weights (W1t[c]: 1024 x 16B, W2t[c]: 2048 x 16B) into
// LDS buffer sdst (W1 at elems [0,8192), W2 at [8192,24576)).
__device__ __forceinline__ void stage_weights(unsigned short* sdst,
                                              const unsigned short* W1c,
                                              const unsigned short* W2c,
                                              int tid) {
#if ASYNC_STAGE
#pragma unroll
    for (int i = 0; i < 12; ++i) {
        int idx = tid + i * 256;
        unsigned short* g = const_cast<unsigned short*>(
            (idx < 1024) ? (W1c + idx * 8) : (W2c + (idx - 1024) * 8));
        __builtin_amdgcn_global_load_async_to_lds_b128(
            GLOBAL_V4I(g), LDS_V4I(sdst + idx * 8), 0, 0);
    }
#else
    const uint4* g1 = (const uint4*)W1c;   // 1024 uint4
    const uint4* g2 = (const uint4*)W2c;   // 2048 uint4
    uint4* s = (uint4*)sdst;
#pragma unroll
    for (int i = 0; i < 12; ++i) {
        int idx = tid + i * 256;
        s[idx] = (idx < 1024) ? g1[idx] : g2[idx - 1024];
    }
#endif
}

__device__ __forceinline__ void wait_stage() {
#if ASYNC_STAGE
#if __has_builtin(__builtin_amdgcn_s_wait_asynccnt)
    __builtin_amdgcn_s_wait_asynccnt(0);
#else
    asm volatile("s_wait_asynccnt 0" ::: "memory");
#endif
#endif
}

// ---------------------------------------------------------------------------
// Prep kernels: transpose weights into f16 workspace
//   W1t[c][h][f]  = f16(W1[c][f][h])      (32,128,64)
//   W2t[c][k][h]  = f16(W2[c][h][k])      (32,128,128)
//   Wzt[o][h]     = f16(Wz[h][o])         (64,128)
//   b2m[h]        = mean_c b2[c][h]       (128)
// ---------------------------------------------------------------------------
__global__ void prep_w1(const float* __restrict__ W1, unsigned short* __restrict__ W1t) {
    int i = blockIdx.x * 256 + threadIdx.x;            // 262144
    if (i >= N_CASC * N_HID * N_IN) return;
    int f = i & 63, h = (i >> 6) & 127, c = i >> 13;
    W1t[i] = f2h(W1[(c << 13) + f * N_HID + h]);
}

__global__ void prep_w2(const float* __restrict__ W2, unsigned short* __restrict__ W2t) {
    int i = blockIdx.x * 256 + threadIdx.x;            // 524288
    if (i >= N_CASC * N_HID * N_HID) return;
    int ki = i & 127, ko = (i >> 7) & 127, c = i >> 14;
    W2t[i] = f2h(W2[(c << 14) + ki * N_HID + ko]);
}

__global__ void prep_wz(const float* __restrict__ Wz, unsigned short* __restrict__ Wzt) {
    int i = blockIdx.x * 256 + threadIdx.x;            // 8192
    if (i >= N_OUT * N_HID) return;
    int h = i & 127, o = i >> 7;
    Wzt[i] = f2h(Wz[h * N_OUT + o]);
}

__global__ void prep_b2(const float* __restrict__ b2, float* __restrict__ b2m) {
    int h = threadIdx.x;                                // 128
    float s = 0.f;
    for (int c = 0; c < N_CASC; ++c) s += b2[c * N_HID + h];
    b2m[h] = s * (1.f / (float)N_CASC);
}

// ---------------------------------------------------------------------------
// Main fused kernel (transposed formulation, lane = node column):
//   Htile = relu(W1t[c] @ Xt + b1)      GEMM1, per cascade
//   Y    += W2t[c] @ Htile              GEMM2 accumulated across cascades
//   Y     = relu(Y/32 + b2mean)
//   Z     = relu(Wzt @ Y + bz)          head GEMM
// Each wave: 16 nodes. 8 waves per block -> 128 nodes/block.
// Weights double-buffered in LDS; next cascade staged asynchronously.
// ---------------------------------------------------------------------------
__global__ __launch_bounds__(256) void enc_main(
    const float* __restrict__ in,               // [N][32][64]
    const float* __restrict__ b1,               // [32][128]
    const unsigned short* __restrict__ W1t,     // [32][128][64]  f16
    const unsigned short* __restrict__ W2t,     // [32][128][128] f16
    const unsigned short* __restrict__ Wzt,     // [64][128]      f16
    const float* __restrict__ b2m,              // [128]
    const float* __restrict__ bz,               // [64]
    float* __restrict__ out)                    // [N][64]
{
    // dynamic LDS: 2 buffers x (8192 + 16384) f16 = 96 KB
    extern __shared__ unsigned short smem[];
    unsigned short* sbuf0 = smem;
    unsigned short* sbuf1 = smem + 24576;

    const int tid     = threadIdx.x;
    const int wave    = tid >> 5;
    const int lane    = tid & 31;
    const int halfsel = lane >> 4;     // 0: lanes 0-15, 1: lanes 16-31
    const int lm      = lane & 15;

    const int  node_raw = blockIdx.x * 128 + wave * 16 + lm;
    const bool valid    = node_raw < N_NODES;
    const int  node     = valid ? node_raw : (N_NODES - 1);
    const float* inrow  = in + (size_t)node * (N_CASC * N_IN);

    // warm the head weights into cache for the epilogue (global_prefetch_b8)
    __builtin_prefetch((const char*)Wzt + tid * 64, 0, 0);

    // Y accumulator: 8 tiles of [16 hid x 16 node] f32
    v8f Y[8];
#pragma unroll
    for (int t = 0; t < 8; ++t) Y[t] = 0.f;

    // kick off staging of cascade 0
    stage_weights(sbuf0, W1t, W2t, tid);

    for (int c = 0; c < N_CASC; ++c) {
        wait_stage();          // this wave's async copies landed in LDS
        __syncthreads();       // all waves' copies landed; prev buffer free

        // issue next cascade's staging into the other buffer (overlaps compute)
        if (c + 1 < N_CASC)
            stage_weights((c & 1) ? sbuf0 : sbuf1,
                          W1t + (c + 1) * 8192, W2t + (c + 1) * 16384, tid);

        const unsigned short* sw = (c & 1) ? sbuf1 : sbuf0;

        // ---- build B fragments of Xt (32x16, f16) for the 2 k-steps ---
        // B layout: lanes 0-15 hold K=0..15, lanes 16-31 hold K=16..31.
        uint4 xq[2][2];
#pragma unroll
        for (int ks = 0; ks < 2; ++ks) {
            const float* p = inrow + c * N_IN + ks * 32 + halfsel * 16;
            float4 a0 = ((const float4*)p)[0];
            float4 a1 = ((const float4*)p)[1];
            float4 a2 = ((const float4*)p)[2];
            float4 a3 = ((const float4*)p)[3];
            xq[ks][0] = make_uint4(pkh(a0.x, a0.y), pkh(a0.z, a0.w),
                                   pkh(a1.x, a1.y), pkh(a1.z, a1.w));
            xq[ks][1] = make_uint4(pkh(a2.x, a2.y), pkh(a2.z, a2.w),
                                   pkh(a3.x, a3.y), pkh(a3.z, a3.w));
        }

        // ---- GEMM1: Htile[t] = b1 + W1t_tile @ Xt ---------------------
        v8f H[8];
#pragma unroll
        for (int t = 0; t < 8; ++t) {
            const float* bp = b1 + c * N_HID + t * 16 + 8 * halfsel;
            float4 q0 = ((const float4*)bp)[0];
            float4 q1 = ((const float4*)bp)[1];
            v8f Cacc;
            Cacc[0] = q0.x; Cacc[1] = q0.y; Cacc[2] = q0.z; Cacc[3] = q0.w;
            Cacc[4] = q1.x; Cacc[5] = q1.y; Cacc[6] = q1.z; Cacc[7] = q1.w;
#pragma unroll
            for (int ks = 0; ks < 2; ++ks) {
                int off = (t * 16 + lm) * N_IN + ks * 32 + halfsel * 16;
                Frag A;
                A.q[0] = *(const uint4*)(sw + off);
                A.q[1] = *(const uint4*)(sw + off + 8);
                Frag B;
                B.q[0] = xq[ks][0];
                B.q[1] = xq[ks][1];
                Cacc = __builtin_amdgcn_wmma_f32_16x16x32_f16(
                    false, A.v, false, B.v, (short)0, Cacc, false, false);
            }
            H[t] = Cacc;
        }

        // ---- GEMM2: Y += W2t_tile @ relu(H) ---------------------------
#pragma unroll
        for (int ks = 0; ks < 4; ++ks) {
            // B frag from H[2ks], H[2ks+1]: relu, pack f16, half-wave swap
            unsigned xp[4], yp[4];
#pragma unroll
            for (int j = 0; j < 4; ++j) {
                xp[j] = pkh(fmaxf(H[2 * ks][2 * j], 0.f),
                            fmaxf(H[2 * ks][2 * j + 1], 0.f));
                yp[j] = pkh(fmaxf(H[2 * ks + 1][2 * j], 0.f),
                            fmaxf(H[2 * ks + 1][2 * j + 1], 0.f));
            }
            Frag B;
#pragma unroll
            for (int j = 0; j < 4; ++j) {
                unsigned xs = (unsigned)__shfl_xor((int)xp[j], 16, 32);
                unsigned ys = (unsigned)__shfl_xor((int)yp[j], 16, 32);
                ((unsigned*)&B.q[0])[j] = halfsel ? ys : xp[j];
                ((unsigned*)&B.q[1])[j] = halfsel ? yp[j] : xs;
            }
#pragma unroll
            for (int t = 0; t < 8; ++t) {
                int off = 8192 + (t * 16 + lm) * N_HID + ks * 32 + halfsel * 16;
                Frag A;
                A.q[0] = *(const uint4*)(sw + off);
                A.q[1] = *(const uint4*)(sw + off + 8);
                Y[t] = __builtin_amdgcn_wmma_f32_16x16x32_f16(
                    false, A.v, false, B.v, (short)0, Y[t], false, false);
            }
        }
        // no trailing barrier: next iteration's wait+barrier orders reuse
    }

    // ---- finalize Y: mean over cascades + b2mean, relu ----------------
#pragma unroll
    for (int t = 0; t < 8; ++t) {
        const float* bp = b2m + t * 16 + 8 * halfsel;
        float4 m0 = ((const float4*)bp)[0];
        float4 m1 = ((const float4*)bp)[1];
        float mm[8] = { m0.x, m0.y, m0.z, m0.w, m1.x, m1.y, m1.z, m1.w };
#pragma unroll
        for (int j = 0; j < 8; ++j)
            Y[t][j] = fmaxf(Y[t][j] * (1.f / (float)N_CASC) + mm[j], 0.f);
    }

    // ---- head GEMM: Z = relu(Wzt @ Y + bz) ----------------------------
    v8f Z[4];
#pragma unroll
    for (int t = 0; t < 4; ++t) {
        const float* bp = bz + t * 16 + 8 * halfsel;
        float4 q0 = ((const float4*)bp)[0];
        float4 q1 = ((const float4*)bp)[1];
        Z[t][0] = q0.x; Z[t][1] = q0.y; Z[t][2] = q0.z; Z[t][3] = q0.w;
        Z[t][4] = q1.x; Z[t][5] = q1.y; Z[t][6] = q1.z; Z[t][7] = q1.w;
    }
#pragma unroll
    for (int ks = 0; ks < 4; ++ks) {
        unsigned xp[4], yp[4];
#pragma unroll
        for (int j = 0; j < 4; ++j) {
            xp[j] = pkh(Y[2 * ks][2 * j],     Y[2 * ks][2 * j + 1]);
            yp[j] = pkh(Y[2 * ks + 1][2 * j], Y[2 * ks + 1][2 * j + 1]);
        }
        Frag B;
#pragma unroll
        for (int j = 0; j < 4; ++j) {
            unsigned xs = (unsigned)__shfl_xor((int)xp[j], 16, 32);
            unsigned ys = (unsigned)__shfl_xor((int)yp[j], 16, 32);
            ((unsigned*)&B.q[0])[j] = halfsel ? ys : xp[j];
            ((unsigned*)&B.q[1])[j] = halfsel ? yp[j] : xs;
        }
#pragma unroll
        for (int t = 0; t < 4; ++t) {
            int off = (t * 16 + lm) * N_HID + ks * 32 + halfsel * 16;
            Frag A;
            A.q[0] = *(const uint4*)(Wzt + off);
            A.q[1] = *(const uint4*)(Wzt + off + 8);
            Z[t] = __builtin_amdgcn_wmma_f32_16x16x32_f16(
                false, A.v, false, B.v, (short)0, Z[t], false, false);
        }
    }

    // ---- store: lane = node, VGPR j = consecutive out features --------
    if (valid) {
        float* op = out + (size_t)node_raw * N_OUT;
#pragma unroll
        for (int t = 0; t < 4; ++t) {
            float4 r0, r1;
            r0.x = fmaxf(Z[t][0], 0.f); r0.y = fmaxf(Z[t][1], 0.f);
            r0.z = fmaxf(Z[t][2], 0.f); r0.w = fmaxf(Z[t][3], 0.f);
            r1.x = fmaxf(Z[t][4], 0.f); r1.y = fmaxf(Z[t][5], 0.f);
            r1.z = fmaxf(Z[t][6], 0.f); r1.w = fmaxf(Z[t][7], 0.f);
            float4* dst = (float4*)(op + t * 16 + 8 * halfsel);
            dst[0] = r0;
            dst[1] = r1;
        }
    }
}

// ---------------------------------------------------------------------------
// Launch
// ---------------------------------------------------------------------------
extern "C" void kernel_launch(void* const* d_in, const int* in_sizes, int n_in,
                              void* d_out, int out_size, void* d_ws, size_t ws_size,
                              hipStream_t stream) {
    const float* in = (const float*)d_in[0];
    const float* W1 = (const float*)d_in[1];
    const float* b1 = (const float*)d_in[2];
    const float* W2 = (const float*)d_in[3];
    const float* b2 = (const float*)d_in[4];
    const float* Wz = (const float*)d_in[5];
    const float* bz = (const float*)d_in[6];

    char* ws = (char*)d_ws;
    unsigned short* W1t = (unsigned short*)(ws);                       // 512 KB
    unsigned short* W2t = (unsigned short*)(ws + 524288);              // 1 MB
    unsigned short* Wzt = (unsigned short*)(ws + 524288 + 1048576);    // 16 KB
    float*          b2m = (float*)(ws + 524288 + 1048576 + 16384);     // 512 B

    prep_w1<<<1024, 256, 0, stream>>>(W1, W1t);
    prep_w2<<<2048, 256, 0, stream>>>(W2, W2t);
    prep_wz<<<32,   256, 0, stream>>>(Wz, Wzt);
    prep_b2<<<1,    128, 0, stream>>>(b2, b2m);

    enc_main<<<(N_NODES + 127) / 128, 256, 96 * 1024, stream>>>(
        in, b1, W1t, W2t, Wzt, b2m, bz, (float*)d_out);
}
